// MultiHeadAttention_61151744360499
// MI455X (gfx1250) — compile-verified, matrix operand order checked by reading, not run on hardware
//
#include <hip/hip_runtime.h>
#include <stdint.h>

// Problem constants (from reference): B=2, S=2048, D=1024, H=16, DK=64
#define B_  2
#define S_  2048
#define D_  1024
#define H_  16
#define DK_ 64

typedef __attribute__((ext_vector_type(16))) __bf16 v16bf;
typedef __attribute__((ext_vector_type(8)))  float  v8f;

union FragU { v16bf v; uint4 q[2]; };

__device__ __forceinline__ uint16_t f2bf(float f) {
    union { float f; uint32_t u; } c; c.f = f;
    uint32_t u = c.u;
    u += 0x7fffu + ((u >> 16) & 1u);   // round-to-nearest-even
    return (uint16_t)(u >> 16);
}

// Load one 16-bit WMMA A/B fragment: two contiguous 16-byte chunks.
// Caller passes p already offset by the lane's K-half (0 or 8 halves);
// halves [0..7] = K{base..base+7}, halves [8..15] = K{base+16..base+23}.
__device__ __forceinline__ v16bf load_frag(const uint16_t* p) {
    FragU f;
    f.q[0] = *(const uint4*)(p);
    f.q[1] = *(const uint4*)(p + 16);
    return f.v;
}

__device__ __forceinline__ v8f zero8() {
    v8f z = {0.f, 0.f, 0.f, 0.f, 0.f, 0.f, 0.f, 0.f};
    return z;
}

// ---------------------------------------------------------------------------
// f32 -> bf16 straight conversion (x staging)
// ---------------------------------------------------------------------------
__global__ __launch_bounds__(256)
void cvt_bf16_k(const float* __restrict__ in, uint16_t* __restrict__ out, int n) {
    int i = (blockIdx.x * 256 + threadIdx.x) * 4;
    if (i + 3 < n) {
        float4 f = *(const float4*)(in + i);
        out[i + 0] = f2bf(f.x);
        out[i + 1] = f2bf(f.y);
        out[i + 2] = f2bf(f.z);
        out[i + 3] = f2bf(f.w);
    }
}

// ---------------------------------------------------------------------------
// f32 W (K x N, row-major) -> bf16 W^T (N x K, row-major) so B-fragments
// (one weight column per lane) are contiguous loads.
// ---------------------------------------------------------------------------
__global__ __launch_bounds__(256)
void cvt_transpose_k(const float* __restrict__ in, uint16_t* __restrict__ out) {
    int i = blockIdx.x * 256 + threadIdx.x;        // 0 .. 1M-1
    int n = i & (D_ - 1);
    int k = i >> 10;
    out[(size_t)n * D_ + k] = f2bf(in[(size_t)k * D_ + n]);
}

// ---------------------------------------------------------------------------
// GEMM: C(4096 x 1024) = A_bf16 @ W (via W^T rows) + bias
// MODE 0: store bf16 head-split   (B,H,S,DK)   (used for Q, K)
// MODE 1: store bf16 head-split-T (B,H,DK,S)   (used for V)
// MODE 2: store f32 row-major     (B,S,D)      (final output)
// Block = 256 threads = 8 waves; wave tile 32x128; block tile 64x512.
// Per 32-K step: 10 fragment loads feed 16 WMMAs (A reuse x8, B reuse x2).
// ---------------------------------------------------------------------------
template <int MODE>
__global__ __launch_bounds__(256)
void gemm_bf16_k(const uint16_t* __restrict__ A,
                 const uint16_t* __restrict__ Wt,
                 const float*    __restrict__ bias,
                 uint16_t*       __restrict__ obf,
                 float*          __restrict__ of32) {
    const int lane   = threadIdx.x & 31;
    const int wave   = threadIdx.x >> 5;
    const int lane16 = lane & 15;
    const int khalf  = (lane & 16) ? 8 : 0;   // K-half select for A/B fragments
    const int mh     = (lane & 16) ? 8 : 0;   // row-half select for C layout
    const int row0   = blockIdx.y * 64 + (wave & 1) * 32;
    const int col0   = blockIdx.x * 512 + (wave >> 1) * 128;

    v8f acc[2][8];
#pragma unroll
    for (int i = 0; i < 2; ++i)
#pragma unroll
        for (int j = 0; j < 8; ++j) acc[i][j] = zero8();

    const uint16_t* ap0 = A  + (size_t)(row0 + lane16) * D_ + khalf;
    const uint16_t* bp0 = Wt + (size_t)(col0 + lane16) * D_ + khalf;

    for (int kk = 0; kk < D_; kk += 32) {
        // Prefetch next K-slice into the cache hierarchy (global_prefetch_b8).
        if (kk + 32 < D_) {
#pragma unroll
            for (int im = 0; im < 2; ++im)
                __builtin_prefetch(ap0 + (size_t)im * 16 * D_ + kk + 32, 0, 1);
#pragma unroll
            for (int in = 0; in < 8; ++in)
                __builtin_prefetch(bp0 + (size_t)in * 16 * D_ + kk + 32, 0, 1);
        }
        v16bf af[2], bfr[8];
#pragma unroll
        for (int im = 0; im < 2; ++im)
            af[im] = load_frag(ap0 + (size_t)im * 16 * D_ + kk);
#pragma unroll
        for (int in = 0; in < 8; ++in)
            bfr[in] = load_frag(bp0 + (size_t)in * 16 * D_ + kk);
#pragma unroll
        for (int im = 0; im < 2; ++im)
#pragma unroll
            for (int in = 0; in < 8; ++in)
                acc[im][in] = __builtin_amdgcn_wmma_f32_16x16x32_bf16(
                    false, af[im], false, bfr[in], (short)0, acc[im][in], false, false);
    }

#pragma unroll
    for (int im = 0; im < 2; ++im) {
#pragma unroll
        for (int in = 0; in < 8; ++in) {
            const int   n  = col0 + in * 16 + lane16;
            const float bv = bias[n];
#pragma unroll
            for (int r = 0; r < 8; ++r) {
                const int   m   = row0 + im * 16 + r + mh;
                const float val = acc[im][in][r] + bv;
                if (MODE == 0) {
                    int b = m >> 11, s = m & (S_ - 1), h = n >> 6, dk = n & (DK_ - 1);
                    obf[(((size_t)(b * H_ + h)) * S_ + s) * DK_ + dk] = f2bf(val);
                } else if (MODE == 1) {
                    int b = m >> 11, s = m & (S_ - 1), h = n >> 6, dk = n & (DK_ - 1);
                    obf[(((size_t)(b * H_ + h)) * DK_ + dk) * S_ + s] = f2bf(val);
                } else {
                    of32[(size_t)m * D_ + n] = val;
                }
            }
        }
    }
}

// ---------------------------------------------------------------------------
// Causal flash attention. One wave = 16 query rows; key tiles of 32.
// Q,K in (B,H,S,DK) bf16; V transposed (B,H,DK,S) bf16.
// Output: (B,S,D) bf16 (pre-Wo). Scores never touch HBM (536 MB > 192 MB L2).
// ---------------------------------------------------------------------------
__global__ __launch_bounds__(256)
void flash_attn_k(const uint16_t* __restrict__ Q,
                  const uint16_t* __restrict__ Kb,
                  const uint16_t* __restrict__ Vt,
                  uint16_t*       __restrict__ O) {
    // Per-wave P staging: 16 rows x 32 cols, row stride 48 halves (96B, 16B-aligned)
    __shared__ __align__(16) uint16_t pshm[8][16 * 48];

    const int lane   = threadIdx.x & 31;
    const int wave   = threadIdx.x >> 5;
    const int lane16 = lane & 15;
    const int khalf  = (lane & 16) ? 8 : 0;
    const int mh     = khalf;                 // C-layout row-half offset
    const int bh     = blockIdx.y;            // b*H + h
    const int qbase  = blockIdx.x * 128 + wave * 16;

    // Q fragments for this wave's 16 rows (kdim 0..31 and 32..63), kept in VGPRs.
    const uint16_t* qrow = Q + ((size_t)bh * S_ + qbase + lane16) * DK_ + khalf;
    const v16bf aq0 = load_frag(qrow);
    const v16bf aq1 = load_frag(qrow + 32);

    float mrun[8], lrun[8];
#pragma unroll
    for (int r = 0; r < 8; ++r) { mrun[r] = -1e30f; lrun[r] = 0.f; }
    v8f o[4];
#pragma unroll
    for (int nc = 0; nc < 4; ++nc) o[nc] = zero8();

    uint16_t* pw = &pshm[wave][0];
    const int ntiles = (qbase + 47) >> 5;     // cover keys 0 .. qbase+15 (causal)

    for (int j = 0; j < ntiles; ++j) {
        const int j0 = j * 32;

        // S = Q @ K^T : B-fragment lane = key column, contiguous over kdim.
        const uint16_t* kr = Kb + ((size_t)bh * S_ + j0 + lane16) * DK_ + khalf;
        v16bf bk00 = load_frag(kr);
        v16bf bk01 = load_frag(kr + 32);
        v16bf bk10 = load_frag(kr + 16 * DK_);
        v16bf bk11 = load_frag(kr + 16 * DK_ + 32);
        v8f c0 = zero8(), c1 = zero8();
        c0 = __builtin_amdgcn_wmma_f32_16x16x32_bf16(false, aq0, false, bk00, (short)0, c0, false, false);
        c0 = __builtin_amdgcn_wmma_f32_16x16x32_bf16(false, aq1, false, bk01, (short)0, c0, false, false);
        c1 = __builtin_amdgcn_wmma_f32_16x16x32_bf16(false, aq0, false, bk10, (short)0, c1, false, false);
        c1 = __builtin_amdgcn_wmma_f32_16x16x32_bf16(false, aq1, false, bk11, (short)0, c1, false, false);

        // Online softmax: row stats reduced across the 16-lane half-wave that
        // holds each row (C layout: lanes 0-15 -> M=r, lanes 16-31 -> M=r+8).
        float alpha[8];
#pragma unroll
        for (int r = 0; r < 8; ++r) {
            const int sq = qbase + r + mh;
            const int kt = j0 + lane16;
            float s0 = (kt      <= sq) ? c0[r] * 0.125f : -1e30f;  // 1/sqrt(64)
            float s1 = (kt + 16 <= sq) ? c1[r] * 0.125f : -1e30f;
            float mx = fmaxf(s0, s1);
            mx = fmaxf(mx, __shfl_xor(mx, 1));
            mx = fmaxf(mx, __shfl_xor(mx, 2));
            mx = fmaxf(mx, __shfl_xor(mx, 4));
            mx = fmaxf(mx, __shfl_xor(mx, 8));
            const float mnew = fmaxf(mrun[r], mx);
            const float al   = __expf(mrun[r] - mnew);
            const float p0   = __expf(s0 - mnew);
            const float p1   = __expf(s1 - mnew);
            float ls = p0 + p1;
            ls += __shfl_xor(ls, 1);
            ls += __shfl_xor(ls, 2);
            ls += __shfl_xor(ls, 4);
            ls += __shfl_xor(ls, 8);
            lrun[r]  = lrun[r] * al + ls;
            mrun[r]  = mnew;
            alpha[r] = al;
            // C layout -> LDS (16x32 bf16 P tile)
            pw[(r + mh) * 48 + lane16]      = f2bf(p0);
            pw[(r + mh) * 48 + 16 + lane16] = f2bf(p1);
        }
        // Same-wave cross-lane LDS RAW: wait for our ds_stores before ds_loads.
        asm volatile("s_wait_dscnt 0" ::: "memory");
        const v16bf ap = load_frag(pw + lane16 * 48 + khalf);  // A layout of P

        // O = O*alpha + P @ V (V transposed -> contiguous B-fragment loads)
#pragma unroll
        for (int nc = 0; nc < 4; ++nc) {
#pragma unroll
            for (int r = 0; r < 8; ++r) o[nc][r] *= alpha[r];
            const v16bf bv = load_frag(
                Vt + ((size_t)bh * DK_ + nc * 16 + lane16) * S_ + j0 + khalf);
            o[nc] = __builtin_amdgcn_wmma_f32_16x16x32_bf16(
                false, ap, false, bv, (short)0, o[nc], false, false);
        }
    }

    // Normalize and store merged-head bf16 output (B,S,D).
    const int b = bh >> 4, h = bh & 15;
#pragma unroll
    for (int nc = 0; nc < 4; ++nc) {
#pragma unroll
        for (int r = 0; r < 8; ++r) {
            const int sq = qbase + r + mh;
            O[((size_t)(b * S_ + sq)) * D_ + h * DK_ + nc * 16 + lane16] =
                f2bf(o[nc][r] / lrun[r]);
        }
    }
}

// ---------------------------------------------------------------------------
// Host launcher
// ---------------------------------------------------------------------------
extern "C" void kernel_launch(void* const* d_in, const int* in_sizes, int n_in,
                              void* d_out, int out_size, void* d_ws, size_t ws_size,
                              hipStream_t stream) {
    (void)in_sizes; (void)n_in; (void)out_size; (void)ws_size;
    const float* x  = (const float*)d_in[0];
    // d_in[1] = mask (causal; implemented analytically)
    const float* Wq = (const float*)d_in[2];
    const float* bq = (const float*)d_in[3];
    const float* Wk = (const float*)d_in[4];
    const float* bk = (const float*)d_in[5];
    const float* Wv = (const float*)d_in[6];
    const float* bv = (const float*)d_in[7];
    const float* Wo = (const float*)d_in[8];
    const float* bo = (const float*)d_in[9];
    float* out = (float*)d_out;

    // bf16 staging carved from workspace (~48 MB total)
    const size_t MD = (size_t)B_ * S_ * D_;   // 4 Mi elements
    const size_t DD = (size_t)D_ * D_;        // 1 Mi elements
    uint16_t* xb  = (uint16_t*)d_ws;
    uint16_t* wqT = xb  + MD;
    uint16_t* wkT = wqT + DD;
    uint16_t* wvT = wkT + DD;
    uint16_t* woT = wvT + DD;
    uint16_t* qb  = woT + DD;
    uint16_t* kb  = qb  + MD;
    uint16_t* vt  = kb  + MD;
    uint16_t* ab  = vt  + MD;

    // 1) precision staging
    cvt_bf16_k<<<(int)(MD / (256 * 4)), 256, 0, stream>>>(x, xb, (int)MD);
    cvt_transpose_k<<<(int)(DD / 256), 256, 0, stream>>>(Wq, wqT);
    cvt_transpose_k<<<(int)(DD / 256), 256, 0, stream>>>(Wk, wkT);
    cvt_transpose_k<<<(int)(DD / 256), 256, 0, stream>>>(Wv, wvT);
    cvt_transpose_k<<<(int)(DD / 256), 256, 0, stream>>>(Wo, woT);

    // 2) Q/K/V projections (WMMA GEMM); block tile 64x512
    dim3 gg(D_ / 512, (B_ * S_) / 64), bb(256);
    gemm_bf16_k<0><<<gg, bb, 0, stream>>>(xb, wqT, bq, qb, nullptr);
    gemm_bf16_k<0><<<gg, bb, 0, stream>>>(xb, wkT, bk, kb, nullptr);
    gemm_bf16_k<1><<<gg, bb, 0, stream>>>(xb, wvT, bv, vt, nullptr);

    // 3) causal flash attention
    flash_attn_k<<<dim3(S_ / 128, B_ * H_), 256, 0, stream>>>(qb, kb, vt, ab);

    // 4) output projection (f32 result)
    gemm_bf16_k<2><<<gg, bb, 0, stream>>>(ab, woT, bo, nullptr, out);
}
